// HierarchicalSoftmax_38165079392655
// MI455X (gfx1250) — matrix operand order repaired and checked by reading, not compiled, not run on hardware
//
#include <hip/hip_runtime.h>
#include <hip/hip_bf16.h>

typedef __attribute__((ext_vector_type(16))) _Float16 v16h;
typedef __attribute__((ext_vector_type(8)))  _Float16 v8h;
typedef __attribute__((ext_vector_type(8)))  float    v8f;

#define B_TOT    131072
#define D_IN     300
#define D_H      512
#define DEPTH    17
#define K_PAD    320          // 10 K-steps of 32 (300 zero-padded)
#define K_STEPS  10
#define LDA      328          // LDS A row stride in halfs (16B-aligned, bank-spread)
#define M_TILE   64           // rows per workgroup
#define FRAG_HALFS 512        // one 32x16 f16 B fragment = 512 halfs
#define W1H_HALFS (K_STEPS * 32 * FRAG_HALFS)   // 163840 halfs = 320KB in d_ws

// ---- dynamic LDS layout (bytes) ----
#define AH_OFF   0
#define AH_BYTES (M_TILE * LDA * 2)        // 41984
#define BS_BYTES (32 * D_H * 2)            // 32768 per buffer
#define BS0_OFF  AH_BYTES                  // 41984
#define BS1_OFF  (BS0_OFF + BS_BYTES)      // 74752
#define B1_OFF   (BS1_OFF + BS_BYTES)      // 107520
#define W2_OFF   (B1_OFF + D_H * 4)        // 109568
#define RED_OFF  (W2_OFF + 2 * D_H * 4)    // 113664
#define SMEM_BYTES (RED_OFF + M_TILE * 2 * 2 * 4)   // 114688

#if defined(__has_builtin)
#if __has_builtin(__builtin_amdgcn_sched_group_barrier)
#define SCHED_GROUP(mask, size, id) __builtin_amdgcn_sched_group_barrier(mask, size, id)
#endif
#endif
#ifndef SCHED_GROUP
#define SCHED_GROUP(mask, size, id)
#endif

// ---------------------------------------------------------------------------
// Pack W1 [512,300] f32 -> f16 WMMA B-fragments in d_ws.
// B 32x16 layout used here (contiguous k, same as A): lanes 0-15 hold K=0..15 (half i <-> K=i),
// lanes 16-31 hold K=16..31 (half i <-> K=16+i); column N = lane%16.
// Fragment f = kt*32 + nt stored contiguously: halfs [f*512 .. f*512+512).
// ---------------------------------------------------------------------------
__global__ void hsm_pack_w1(const float* __restrict__ W1, _Float16* __restrict__ w1h) {
    int idx = blockIdx.x * 256 + threadIdx.x;
    if (idx >= W1H_HALFS) return;
    int i    = idx & 15;
    int lane = (idx >> 4) & 31;
    int f    = idx >> 9;
    int kt   = f >> 5;
    int nt   = f & 31;
    int k = kt * 32 + (lane >> 4) * 16 + i;
    int n = nt * 16 + (lane & 15);
    float v = (k < D_IN) ? W1[n * D_IN + k] : 0.f;
    w1h[idx] = (_Float16)v;
}

// ---------------------------------------------------------------------------
// Main fused kernel: GEMM1 (WMMA f16 -> f32) + bias + sigmoid + 512x2 head
// + path log-prob sum. 64 rows per WG, 8 waves = 4 M-subtiles x 2 N-halves.
// B fragments double-buffered in LDS. sched_group_barrier forces the machine
// scheduler to front-load 4 B fragments and then interleave WMMA / DS-reads,
// so WMMAs issue under partial s_wait_dscnt instead of full drains.
// ---------------------------------------------------------------------------
__global__ void __launch_bounds__(256) hsm_main(
    const float* __restrict__ wv,   // [B, 300]
    const int*   __restrict__ ps,   // [B, 17]
    const _Float16* __restrict__ w1h,
    const float* __restrict__ b1,   // [512]
    const float* __restrict__ W2,   // [2, 512]
    const float* __restrict__ b2,   // [2]
    float* __restrict__ out)        // [B]
{
    extern __shared__ char smem[];
    _Float16* Ah  = (_Float16*)(smem + AH_OFF);
    _Float16* Bs0 = (_Float16*)(smem + BS0_OFF);
    _Float16* Bs1 = (_Float16*)(smem + BS1_OFF);
    float*    b1s = (float*)(smem + B1_OFF);
    float*    w2s = (float*)(smem + W2_OFF);
    float*    red = (float*)(smem + RED_OFF);

    const int tid    = threadIdx.x;
    const int lane   = tid & 31;
    const int wave   = tid >> 5;
    const int m_sub  = wave >> 1;       // 0..3 : which 16-row slab
    const int n_half = wave & 1;        // 0/1  : which 256 hidden cols
    const int laneN  = lane & 15;
    const int laneHi = lane >> 4;
    const int rowBase = m_sub * 16;
    const long gRow0  = (long)blockIdx.x * M_TILE;

    // preload b1 / W2 into LDS
    for (int i = tid; i < D_H; i += 256)      b1s[i] = b1[i];
    for (int i = tid; i < 2 * D_H; i += 256)  w2s[i] = W2[i];

    // A tile: f32 -> f16 into LDS, K zero-padded to 320 (coalesced over k)
    for (int idx = tid; idx < M_TILE * K_PAD; idx += 256) {
        int row = idx / K_PAD;
        int k   = idx - row * K_PAD;
        float v = (k < D_IN) ? wv[(gRow0 + row) * D_IN + k] : 0.f;
        Ah[row * LDA + k] = (_Float16)v;
    }

    // stage B fragments for kt=0 into buffer 0
    {
        const uint4* src = (const uint4*)(w1h);
        uint4*       dst = (uint4*)Bs0;
#pragma unroll
        for (int j = 0; j < 8; ++j) dst[tid + j * 256] = src[tid + j * 256];
    }
    __syncthreads();

    v8f acc[16];
    {
        v8f z = {0.f, 0.f, 0.f, 0.f, 0.f, 0.f, 0.f, 0.f};
#pragma unroll
        for (int t = 0; t < 16; ++t) acc[t] = z;
    }

    for (int kt = 0; kt < K_STEPS; ++kt) {
        _Float16* BsCur = (kt & 1) ? Bs1 : Bs0;
        _Float16* BsNxt = (kt & 1) ? Bs0 : Bs1;

        // Issue next K-step's global loads NOW; latency hides under the WMMAs.
        uint4 stage[8];
        if (kt + 1 < K_STEPS) {
            const uint4* src = (const uint4*)(w1h + (kt + 1) * 32 * FRAG_HALFS);
#pragma unroll
            for (int j = 0; j < 8; ++j) stage[j] = src[tid + j * 256];
        }

        // A fragment (16-bit A 16x32): lane<16 -> K {0..15},
        // lane>=16 -> K {16..31}; two 16B LDS loads.
        const int k0 = kt * 32;
        const _Float16* arow = &Ah[(rowBase + laneN) * LDA + k0 + laneHi * 16];
        v8h alo = *(const v8h*)(arow);
        v8h ahi = *(const v8h*)(arow + 8);
        v16h afrag = __builtin_shufflevector(alo, ahi,
            0, 1, 2, 3, 4, 5, 6, 7, 8, 9, 10, 11, 12, 13, 14, 15);

        // B-fragment pipeline, depth 4.
        const _Float16* bbase = &BsCur[(n_half * 16) * FRAG_HALFS + lane * 16];
        v16h bf[4];
#pragma unroll
        for (int j = 0; j < 4; ++j)
            bf[j] = *(const v16h*)(bbase + j * FRAG_HALFS);

#pragma unroll
        for (int t = 0; t < 16; ++t) {
            acc[t] = __builtin_amdgcn_wmma_f32_16x16x32_f16(
                false, afrag, false, bf[t & 3], (short)0, acc[t], false, false);
            if (t + 4 < 16)
                bf[t & 3] = *(const v16h*)(bbase + (t + 4) * FRAG_HALFS);
        }

        // Pin the schedule: front-load A-frag + 4 B-frags (10 DS reads),
        // then alternate 1 WMMA with the 2 DS reads of the frag 4 tiles
        // ahead, tail of 4 WMMAs. Keeps 4 B buffers live -> partial waits.
        SCHED_GROUP(0x100, 10, 0);          // DS read x10
#pragma unroll
        for (int t = 0; t < 12; ++t) {
            SCHED_GROUP(0x008, 1, 0);       // WMMA x1
            SCHED_GROUP(0x100, 2, 0);       // DS read x2
        }
        SCHED_GROUP(0x008, 4, 0);           // WMMA x4 tail

        // Commit the staged chunk to the other LDS buffer; one barrier per kt.
        if (kt + 1 < K_STEPS) {
            uint4* dst = (uint4*)BsNxt;
#pragma unroll
            for (int j = 0; j < 8; ++j) dst[tid + j * 256] = stage[j];
        }
        __syncthreads();
    }

    // Epilogue on C layout: VGPR r, lanes0-15 -> row r, lanes16-31 -> row 8+r,
    // col = tile*16 + lane%16. Fuse bias+sigmoid+head dot products per lane.
    float p0[8], p1[8];
#pragma unroll
    for (int r = 0; r < 8; ++r) { p0[r] = 0.f; p1[r] = 0.f; }

#pragma unroll
    for (int t = 0; t < 16; ++t) {
        const int n = n_half * 256 + t * 16 + laneN;
        const float bb = b1s[n];
        const float wa = w2s[n];
        const float wb = w2s[D_H + n];
#pragma unroll
        for (int r = 0; r < 8; ++r) {
            float x = acc[t][r] + bb;
            float h = 1.0f / (1.0f + __expf(-x));
            p0[r] += h * wa;
            p1[r] += h * wb;
        }
    }

    // reduce the N dimension across the 16 lanes of each half-wave
#pragma unroll
    for (int r = 0; r < 8; ++r) {
#pragma unroll
        for (int m = 1; m <= 8; m <<= 1) {
            p0[r] += __shfl_xor(p0[r], m, 32);
            p1[r] += __shfl_xor(p1[r], m, 32);
        }
    }

    // lane 0 holds rows rowBase+0..7, lane 16 holds rowBase+8..15
    if (lane == 0 || lane == 16) {
        const int rbase = rowBase + laneHi * 8;
#pragma unroll
        for (int r = 0; r < 8; ++r) {
            red[((rbase + r) * 2 + 0) * 2 + n_half] = p0[r];
            red[((rbase + r) * 2 + 1) * 2 + n_half] = p1[r];
        }
    }
    __syncthreads();

    // final: combine halves, add b2, path log-prob sum via zero-count
    if (tid < M_TILE) {
        const long gRow = gRow0 + tid;
        float o0 = red[(tid * 2 + 0) * 2 + 0] + red[(tid * 2 + 0) * 2 + 1] + b2[0];
        float o1 = red[(tid * 2 + 1) * 2 + 0] + red[(tid * 2 + 1) * 2 + 1] + b2[1];
        int s = 0;
#pragma unroll
        for (int d = 0; d < DEPTH; ++d) s += ps[gRow * DEPTH + d];
        out[gRow] = (float)(DEPTH - s) * __logf(o0) + (float)s * __logf(o1);
    }
}

extern "C" void kernel_launch(void* const* d_in, const int* in_sizes, int n_in,
                              void* d_out, int out_size, void* d_ws, size_t ws_size,
                              hipStream_t stream) {
    const float* wv = (const float*)d_in[0];   // word_vecs [B,300]
    const int*   psd = (const int*)d_in[1];    // path_sides [B,17]
    const float* W1 = (const float*)d_in[2];   // [512,300]
    const float* b1 = (const float*)d_in[3];   // [512]
    const float* W2 = (const float*)d_in[4];   // [2,512]
    const float* b2 = (const float*)d_in[5];   // [2]
    float* out = (float*)d_out;                // [B]
    _Float16* w1h = (_Float16*)d_ws;           // 320KB packed B fragments

    hsm_pack_w1<<<(W1H_HALFS + 255) / 256, 256, 0, stream>>>(W1, w1h);
    hsm_main<<<B_TOT / M_TILE, 256, SMEM_BYTES, stream>>>(wv, psd, w1h, b1, W2, b2, out);
}
